// VGNN_9285719294191
// MI455X (gfx1250) — matrix operand order repaired and checked by reading, not compiled, not run on hardware
//
#include <hip/hip_runtime.h>
#include <hip/hip_bf16.h>

typedef __attribute__((ext_vector_type(16))) _Float16 v16h;
typedef __attribute__((ext_vector_type(8)))  float    v8f;

#define NN   1536
#define FF   128
#define BB   16
#define NDEC 1537
#define SDEC 1552

__device__ __forceinline__ float lrelu(float x){ return x > 0.f ? x : 0.2f*x; }
__device__ __forceinline__ float eluf(float x){ return x > 0.f ? x : expm1f(x); }

__device__ __forceinline__ v8f wmma_f16(v16h a, v16h b, v8f c){
  return __builtin_amdgcn_wmma_f32_16x16x32_f16(false, a, false, b, (short)0, c, false, false);
}

// Build A fragment (16x32 f16) for this lane from an f32 row pointer (row + k0 applied).
// Lane layout: lanes 0-15 K in {0..15} (g0=0); lanes 16-31 K in {16..31} (g0=8).
__device__ __forceinline__ v16h load_a_f32(const float* xr, int g0){
  const float4 x0 = *(const float4*)(xr + 2*g0);
  const float4 x1 = *(const float4*)(xr + 2*g0 + 4);
  const float4 x2 = *(const float4*)(xr + 2*g0 + 8);
  const float4 x3 = *(const float4*)(xr + 2*g0 + 12);
  v16h a;
  a[0]=(_Float16)x0.x;  a[1]=(_Float16)x0.y;  a[2]=(_Float16)x0.z;  a[3]=(_Float16)x0.w;
  a[4]=(_Float16)x1.x;  a[5]=(_Float16)x1.y;  a[6]=(_Float16)x1.z;  a[7]=(_Float16)x1.w;
  a[8]=(_Float16)x2.x;  a[9]=(_Float16)x2.y;  a[10]=(_Float16)x2.z; a[11]=(_Float16)x2.w;
  a[12]=(_Float16)x3.x; a[13]=(_Float16)x3.y; a[14]=(_Float16)x3.z; a[15]=(_Float16)x3.w;
  return a;
}

// ---------------- Weight transpose + f16 convert ----------------
__global__ void k_prep(const float* __restrict__ encW, const float* __restrict__ paramW,
                       const float* __restrict__ decW,
                       _Float16* __restrict__ WT0, _Float16* __restrict__ WT1,
                       _Float16* __restrict__ pWT, _Float16* __restrict__ dWT){
  int i = blockIdx.x*256 + threadIdx.x;
  if (i < 16384){ int n=i>>7, k=i&127; WT0[n*128+k]=(_Float16)encW[k*128+n]; }
  else if (i < 32768){ int j=i-16384; int n=j>>7,k=j&127; WT1[n*128+k]=(_Float16)encW[16384 + k*128+n]; }
  else if (i < 65536){ int j=i-32768; int n=j>>7,k=j&127; pWT[n*128+k]=(_Float16)paramW[k*256+n]; }
  else if (i < 81920){ int j=i-65536; int n=j>>7,k=j&127; dWT[n*128+k]=(_Float16)decW[k*128+n]; }
}

// ---------------- h = X@W (WMMA), writes h^T (f16) + si, sj ----------------
__global__ void k_h(const float* __restrict__ X, unsigned long x_stride,
                    const _Float16* __restrict__ WT,
                    _Float16* __restrict__ hT, unsigned long hT_stride,
                    float* __restrict__ si, float* __restrict__ sj, unsigned long s_stride,
                    const float* __restrict__ a_src, const float* __restrict__ a_dst){
  const int lane = threadIdx.x & 31, wave = threadIdx.x >> 5;
  const int rowbase = (blockIdx.x*4 + wave)*16;
  const int b = blockIdx.y;
  const int m = lane & 15, hi = lane >> 4, g0 = hi ? 8 : 0;
  const float* Xr = X + (unsigned long)b*x_stride + (unsigned long)(rowbase+m)*FF;
  v8f acc[8] = {};
  #pragma unroll
  for (int k0=0;k0<128;k0+=32){
    v16h a = load_a_f32(Xr + k0, g0);
    #pragma unroll
    for (int t=0;t<8;++t){
      const v16h bm = *(const v16h*)(WT + (unsigned long)(16*t+m)*128 + k0 + (hi?16:0));
      acc[t] = wmma_f16(a, bm, acc[t]);
    }
  }
  const int Moff = hi ? 8 : 0;
  float av[8], dv[8];
  #pragma unroll
  for (int t=0;t<8;++t){ av[t]=a_src[16*t+m]; dv[t]=a_dst[16*t+m]; }
  float sip[8], sjp[8];
  #pragma unroll
  for (int v=0;v<8;++v){
    float s1=0.f, s2=0.f;
    #pragma unroll
    for (int t=0;t<8;++t){ float h = acc[t][v]; s1 += h*av[t]; s2 += h*dv[t]; }
    #pragma unroll
    for (int d=1; d<16; d<<=1){ s1 += __shfl_xor(s1, d, 32); s2 += __shfl_xor(s2, d, 32); }
    sip[v]=s1; sjp[v]=s2;
  }
  _Float16* hTb = hT + (unsigned long)b*hT_stride;
  #pragma unroll
  for (int t=0;t<8;++t){
    #pragma unroll
    for (int v=0;v<8;++v)
      hTb[(unsigned long)(16*t+m)*NN + rowbase + Moff + v] = (_Float16)acc[t][v];
  }
  if (m == 0){
    float* sib = si + (unsigned long)b*s_stride;
    float* sjb = sj + (unsigned long)b*s_stride;
    #pragma unroll
    for (int v=0;v<8;++v){ sib[rowbase+Moff+v]=sip[v]; sjb[rowbase+Moff+v]=sjp[v]; }
  }
}

// ---------------- masked sj, sjmax, active count ----------------
__global__ void k_mask(const float* __restrict__ sj, unsigned long sj_stride,
                       const int* __restrict__ data,
                       float* __restrict__ sjm, float* __restrict__ sjmax,
                       float* __restrict__ cnt){
  const int b = blockIdx.x, tid = threadIdx.x;
  const float* sjb = sj + (unsigned long)b*sj_stride;
  __shared__ float rmax[256];
  __shared__ float rcnt[256];
  float mx = -1e30f, c = 0.f;
  for (int k=tid;k<NN;k+=256){
    int g = data[b*NN+k];
    float v = (g != 0) ? sjb[k] : -1e30f;
    sjm[b*NN+k] = v;
    mx = fmaxf(mx, v);
    c += (g==1) ? 1.f : 0.f;
  }
  rmax[tid]=mx; rcnt[tid]=c; __syncthreads();
  for (int s=128;s>0;s>>=1){
    if (tid<s){ rmax[tid]=fmaxf(rmax[tid],rmax[tid+s]); rcnt[tid]+=rcnt[tid+s]; }
    __syncthreads();
  }
  if (tid==0){ sjmax[b]=rmax[0]; if (cnt) cnt[b]=rcnt[0]; }
}

// ---------------- streaming softmax(P) @ h via WMMA, + elu ----------------
__global__ void k_attn(const _Float16* __restrict__ hT, unsigned long hT_stride,
                       const float* __restrict__ si, unsigned long si_stride,
                       const float* __restrict__ sjm,
                       const float* __restrict__ sjmax,
                       const int* __restrict__ data,
                       float* __restrict__ Xout){
  const int lane = threadIdx.x & 31, wave = threadIdx.x >> 5;
  const int rowbase = (blockIdx.x*4 + wave)*16;
  const int b = blockIdx.y;
  const int m = lane & 15, hi = lane >> 4;
  const int row = rowbase + m;
  const float sir = si[(unsigned long)b*si_stride + row];
  const float Mrow = lrelu(sir + sjmax[b]);
  const float* sjb = sjm + (unsigned long)b*NN;
  const unsigned actmask = (unsigned)__ballot(data[b*NN+row] != 0);
  const _Float16* hTb = hT + (unsigned long)b*hT_stride;
  v8f acc[8] = {};
  float rsum = 0.f;
  for (int k0=0;k0<NN;k0+=32){
    const float* sp = sjb + k0 + (hi?16:0);
    const float4 q0 = *(const float4*)(sp);
    const float4 q1 = *(const float4*)(sp+4);
    const float4 q2 = *(const float4*)(sp+8);
    const float4 q3 = *(const float4*)(sp+12);
    float sv[16];
    sv[0]=q0.x; sv[1]=q0.y; sv[2]=q0.z; sv[3]=q0.w;
    sv[4]=q1.x; sv[5]=q1.y; sv[6]=q1.z; sv[7]=q1.w;
    sv[8]=q2.x; sv[9]=q2.y; sv[10]=q2.z; sv[11]=q2.w;
    sv[12]=q3.x; sv[13]=q3.y; sv[14]=q3.z; sv[15]=q3.w;
    v16h a;
    #pragma unroll
    for (int e=0;e<16;++e){
      float p = __expf(lrelu(sir + sv[e]) - Mrow);
      rsum += p;
      a[e] = (_Float16)p;
    }
    #pragma unroll
    for (int t=0;t<8;++t){
      const v16h bm = *(const v16h*)(hTb + (unsigned long)(16*t+m)*NN + k0 + (hi?16:0));
      acc[t] = wmma_f16(a, bm, acc[t]);
    }
  }
  rsum += __shfl_xor(rsum, 16, 32);  // both K-halves of each row
  const int Moff = hi ? 8 : 0;
  float* ob = Xout + (unsigned long)b*NN*FF;
  #pragma unroll
  for (int v=0;v<8;++v){
    float r = __shfl(rsum, v + Moff, 32);
    bool actM = (actmask >> (v + Moff)) & 1u;
    float inv = (actM && r > 0.f) ? 1.f/r : 0.f;
    #pragma unroll
    for (int t=0;t<8;++t){
      float o = acc[t][v] * inv;
      ob[(unsigned long)(rowbase+v+Moff)*FF + 16*t + m] = eluf(o);
    }
  }
}

// ---------------- par = x@param_W + b : mean out + KLD partials ----------------
__global__ void k_param(const float* __restrict__ X,
                        const _Float16* __restrict__ pWT,
                        const float* __restrict__ pb,
                        const int* __restrict__ data,
                        float* __restrict__ mean,
                        float* __restrict__ kldp){
  __shared__ float sred[4];
  const int lane = threadIdx.x & 31, wave = threadIdx.x >> 5;
  const int rowbase = (blockIdx.x*4 + wave)*16;
  const int b = blockIdx.y;
  const int m = lane & 15, hi = lane >> 4, g0 = hi ? 8 : 0;
  const float* Xr = X + (unsigned long)b*NN*FF + (unsigned long)(rowbase+m)*FF;
  v8f acc[16] = {};
  #pragma unroll
  for (int k0=0;k0<128;k0+=32){
    v16h a = load_a_f32(Xr + k0, g0);
    #pragma unroll
    for (int t=0;t<16;++t){
      const v16h bm = *(const v16h*)(pWT + (unsigned long)(16*t+m)*128 + k0 + (hi?16:0));
      acc[t] = wmma_f16(a, bm, acc[t]);
    }
  }
  const int Moff = hi ? 8 : 0;
  const unsigned wmask = (unsigned)__ballot(data[b*NN + rowbase + m] == 1);
  float kp = 0.f;
  float* mb = mean + (unsigned long)b*NN*FF;
  #pragma unroll
  for (int v=0;v<8;++v){
    const bool w = (wmask >> (v+Moff)) & 1u;
    #pragma unroll
    for (int t=0;t<8;++t){
      float mu = acc[t][v]   + pb[16*t+m];
      float sg = acc[t+8][v] + pb[128+16*t+m];
      mb[(unsigned long)(rowbase+v+Moff)*FF + 16*t + m] = mu;
      if (w) kp += __expf(sg) - sg - 1.f + mu*mu;
    }
  }
  #pragma unroll
  for (int d=1;d<32;d<<=1) kp += __shfl_xor(kp, d, 32);
  if (lane==0) sred[wave]=kp;
  __syncthreads();
  if (threadIdx.x==0) kldp[b*24 + blockIdx.x] = sred[0]+sred[1]+sred[2]+sred[3];
}

// ---------------- decoder h = [mean; embed[-1]] @ dec_W (WMMA) ----------------
__global__ void k_hdec(const float* __restrict__ mean, const float* __restrict__ embed,
                       const _Float16* __restrict__ dWT,
                       const float* __restrict__ a_src, const float* __restrict__ a_dst,
                       float* __restrict__ hdec,
                       float* __restrict__ si_d, float* __restrict__ sj_d){
  const int lane = threadIdx.x & 31, wave = threadIdx.x >> 5;
  const int rowbase = (blockIdx.x*4 + wave)*16;
  if (rowbase >= NDEC) return;
  const int b = blockIdx.y;
  const int m = lane & 15, hi = lane >> 4, g0 = hi ? 8 : 0;
  const int row = rowbase + m;
  const float* meanb = mean + (unsigned long)b*NN*FF;
  const float* embedN = embed + (unsigned long)NN*FF;
  v8f acc[8] = {};
  #pragma unroll
  for (int k0=0;k0<128;k0+=32){
    v16h a = {};
    if (row <= NN){
      const float* xr = (row < NN) ? (meanb + (unsigned long)row*FF + k0) : (embedN + k0);
      a = load_a_f32(xr, g0);
    }
    #pragma unroll
    for (int t=0;t<8;++t){
      const v16h bm = *(const v16h*)(dWT + (unsigned long)(16*t+m)*128 + k0 + (hi?16:0));
      acc[t] = wmma_f16(a, bm, acc[t]);
    }
  }
  const int Moff = hi ? 8 : 0;
  float av[8], dv[8];
  #pragma unroll
  for (int t=0;t<8;++t){ av[t]=a_src[16*t+m]; dv[t]=a_dst[16*t+m]; }
  float sip[8], sjp[8];
  #pragma unroll
  for (int v=0;v<8;++v){
    float s1=0.f, s2=0.f;
    #pragma unroll
    for (int t=0;t<8;++t){ float h = acc[t][v]; s1 += h*av[t]; s2 += h*dv[t]; }
    #pragma unroll
    for (int d=1; d<16; d<<=1){ s1 += __shfl_xor(s1, d, 32); s2 += __shfl_xor(s2, d, 32); }
    sip[v]=s1; sjp[v]=s2;
  }
  float* hb = hdec + (unsigned long)b*NDEC*FF;
  #pragma unroll
  for (int t=0;t<8;++t){
    #pragma unroll
    for (int v=0;v<8;++v){
      int r2 = rowbase + v + Moff;
      if (r2 < NDEC) hb[(unsigned long)r2*FF + 16*t + m] = acc[t][v];
    }
  }
  if (m == 0){
    #pragma unroll
    for (int v=0;v<8;++v){
      int r2 = rowbase + v + Moff;
      if (r2 < NDEC){ si_d[b*SDEC + r2] = sip[v]; sj_d[b*SDEC + r2] = sjp[v]; }
    }
  }
}

// ---------------- decoder attention (row N only) + relu ----------------
__global__ void k_dec(const float* __restrict__ hdec,
                      const float* __restrict__ si_d, const float* __restrict__ sj_d,
                      const int* __restrict__ data, float* __restrict__ feats){
  const int b = blockIdx.x, tid = threadIdx.x;
  __shared__ float pl[128];
  __shared__ float red[128];
  const float* sjb = sj_d + b*SDEC;
  const float* hb  = hdec + (unsigned long)b*NDEC*FF;
  const float siN = si_d[b*SDEC + NN];
  // pass 1: max over masked sj (row N is always active)
  float mx = -1e30f;
  for (int j=tid;j<NDEC;j+=128){
    bool a = (j==NN) || (data[b*NN+j] != 0);
    if (a) mx = fmaxf(mx, sjb[j]);
  }
  red[tid]=mx; __syncthreads();
  for (int s=64;s>0;s>>=1){ if (tid<s) red[tid]=fmaxf(red[tid],red[tid+s]); __syncthreads(); }
  const float M = lrelu(siN + red[0]);
  __syncthreads();
  // pass 2: p_j, numerator, denominator
  float accv = 0.f, psum = 0.f;
  for (int j0=0;j0<NDEC;j0+=128){
    int j = j0 + tid;
    float p = 0.f;
    if (j < NDEC){
      bool a = (j==NN) || (data[b*NN+j] != 0);
      p = a ? __expf(lrelu(siN + sjb[j]) - M) : 0.f;
    }
    pl[tid] = p; psum += p;
    __syncthreads();
    int lim = (NDEC - j0 < 128) ? (NDEC - j0) : 128;
    for (int jj=0;jj<lim;++jj) accv += pl[jj] * hb[(unsigned long)(j0+jj)*FF + tid];
    __syncthreads();
  }
  red[tid]=psum; __syncthreads();
  for (int s=64;s>0;s>>=1){ if (tid<s) red[tid]+=red[tid+s]; __syncthreads(); }
  feats[b*FF + tid] = fmaxf(accv/red[0], 0.f);
}

// ---------------- final MLP + KLD sum ----------------
__global__ void k_final(const float* __restrict__ feats,
                        const float* __restrict__ W1, const float* __restrict__ b1,
                        const float* __restrict__ W2, const float* __restrict__ b2,
                        const float* __restrict__ kldp, const float* __restrict__ cnt,
                        float* __restrict__ out){
  __shared__ float t1[16*128];
  const int tid = threadIdx.x;
  for (int i=tid;i<16*128;i+=256){
    int r = i >> 7, c = i & 127;
    float s = b1[c];
    for (int k=0;k<128;++k) s += feats[r*128+k]*W1[k*128+c];
    t1[i] = fmaxf(s, 0.f);
  }
  __syncthreads();
  if (tid < 16){
    float s = b2[0];
    for (int c=0;c<128;++c) s += t1[tid*128+c]*W2[c];
    out[tid] = s;
  }
  if (tid == 0){
    float tot = 0.f;
    for (int b=0;b<16;++b){
      float s = 0.f;
      for (int i=0;i<24;++i) s += kldp[b*24+i];
      tot += 0.5f*s/fmaxf(cnt[b], 1.f);
    }
    out[16] = tot;
  }
}

extern "C" void kernel_launch(void* const* d_in, const int* in_sizes, int n_in,
                              void* d_out, int out_size, void* d_ws, size_t ws_size,
                              hipStream_t stream) {
  const int*   data   = (const int*)d_in[0];
  const float* embed  = (const float*)d_in[1];
  const float* encW   = (const float*)d_in[2];
  const float* e_asrc = (const float*)d_in[3];
  const float* e_adst = (const float*)d_in[4];
  const float* paramW = (const float*)d_in[5];
  const float* paramb = (const float*)d_in[6];
  const float* decW   = (const float*)d_in[7];
  const float* d_asrc = (const float*)d_in[8];
  const float* d_adst = (const float*)d_in[9];
  const float* W1     = (const float*)d_in[10];
  const float* b1     = (const float*)d_in[11];
  const float* W2     = (const float*)d_in[12];
  const float* b2     = (const float*)d_in[13];
  float* out = (float*)d_out;

  char* ws = (char*)d_ws;
  size_t off = 0;
  auto alloc = [&](size_t bytes)->char*{
    char* p = ws + off;
    off = (off + bytes + 255) & ~(size_t)255;
    return p;
  };
  _Float16* WT0  = (_Float16*)alloc((size_t)16384*2);
  _Float16* WT1  = (_Float16*)alloc((size_t)16384*2);
  _Float16* pWT  = (_Float16*)alloc((size_t)32768*2);
  _Float16* dWT  = (_Float16*)alloc((size_t)16384*2);
  _Float16* h0T  = (_Float16*)alloc((size_t)FF*NN*2);
  _Float16* h1T  = (_Float16*)alloc((size_t)BB*FF*NN*2);
  float* si0  = (float*)alloc((size_t)NN*4);
  float* sj0  = (float*)alloc((size_t)NN*4);
  float* si1  = (float*)alloc((size_t)BB*NN*4);
  float* sj1  = (float*)alloc((size_t)BB*NN*4);
  float* sjm  = (float*)alloc((size_t)BB*NN*4);
  float* sjmx = (float*)alloc(64);
  float* cnt  = (float*)alloc(64);
  float* kldp = (float*)alloc((size_t)BB*24*4);
  float* x1   = (float*)alloc((size_t)BB*NN*FF*4);   // reused as x2 (in-place legal)
  float* mean = (float*)alloc((size_t)BB*NN*FF*4);
  float* hdec = (float*)alloc((size_t)BB*NDEC*FF*4);
  float* si_d = (float*)alloc((size_t)BB*SDEC*4);
  float* sj_d = (float*)alloc((size_t)BB*SDEC*4);
  float* feats= (float*)alloc((size_t)BB*FF*4);

  // weights -> transposed f16
  k_prep<<<320,256,0,stream>>>(encW, paramW, decW, WT0, WT1, pWT, dWT);
  // layer 0: h0 is sample-independent (x0 = embed[:-1] for every sample)
  k_h<<<dim3(24,1),128,0,stream>>>(embed, 0ul, WT0, h0T, 0ul, si0, sj0, 0ul, e_asrc, e_adst);
  k_mask<<<BB,256,0,stream>>>(sj0, 0ul, data, sjm, sjmx, cnt);
  k_attn<<<dim3(24,BB),128,0,stream>>>(h0T, 0ul, si0, 0ul, sjm, sjmx, data, x1);
  // layer 1 (per sample)
  k_h<<<dim3(24,BB),128,0,stream>>>(x1, (unsigned long)NN*FF, WT1, h1T,
                                    (unsigned long)FF*NN, si1, sj1, (unsigned long)NN,
                                    e_asrc+FF, e_adst+FF);
  k_mask<<<BB,256,0,stream>>>(sj1, (unsigned long)NN, data, sjm, sjmx, nullptr);
  k_attn<<<dim3(24,BB),128,0,stream>>>(h1T, (unsigned long)FF*NN, si1, (unsigned long)NN,
                                       sjm, sjmx, data, x1);
  // param head: mean + KLD partials
  k_param<<<dim3(24,BB),128,0,stream>>>(x1, pWT, paramb, data, mean, kldp);
  // decoder
  k_hdec<<<dim3(25,BB),128,0,stream>>>(mean, embed, dWT, d_asrc, d_adst, hdec, si_d, sj_d);
  k_dec<<<BB,128,0,stream>>>(hdec, si_d, sj_d, data, feats);
  // output MLP + KLD total
  k_final<<<1,256,0,stream>>>(feats, W1, b1, W2, b2, kldp, cnt, out);

  (void)in_sizes; (void)n_in; (void)out_size; (void)ws_size;
}
